// DynamicFFNLayer_41884521071078
// MI455X (gfx1250) — matrix operand order repaired and checked by reading, not compiled, not run on hardware
//
#include <hip/hip_runtime.h>
#include <hip/hip_bf16.h>
#include <math.h>

// ---------------------------------------------------------------------------
// DynamicFFNLayer for MI455X (gfx1250, wave32, WMMA bf16 16x16x32, f32 acc)
//
// Pipeline:
//   1. ln_cast:   xn = LayerNorm(x) -> bf16 ; also x -> bf16
//   2. cast W1, W2, Wr1 -> bf16
//   3. GEMM<GELU_BF16>:   h      = gelu(xn @ Wr1^T)          [N,D] bf16
//   4. GEMM<F32>:         scores = h  @ W1^T                 [N,F] f32
//   5. topk_mask: per-row exact radix-select of top_k -> u8 mask [N,F]
//   6. GEMM<MASK_GELU>:   s      = gelu(x @ W1^T) * mask     [N,F] bf16
//                         (s overlays the scores region in ws)
//   7. GEMM<F32>:         out    = s @ W2^T                  [N,D] f32
//
// GEMM inner loop: double-buffered LDS fed by GLOBAL_LOAD_ASYNC_TO_LDS_B128
// (ASYNCcnt); XOR-swizzled 16B groups keep b128 alignment and bank-conflict-
// free WMMA fragment reads.  GELU is fully inlined (no libm call) so the
// epilogue keeps all accumulators in VGPRs (no scratch spills).
// ---------------------------------------------------------------------------

typedef __attribute__((ext_vector_type(16))) __bf16 v16bf;
typedef __attribute__((ext_vector_type(8)))  float  v8f;

#if __has_builtin(__builtin_amdgcn_global_load_async_to_lds_b128) && \
    __has_builtin(__builtin_amdgcn_s_wait_asynccnt)
#define USE_ASYNC_LDS 1
#else
#define USE_ASYNC_LDS 0
#endif

#if USE_ASYNC_LDS
typedef int v4i_b128 __attribute__((vector_size(16)));
typedef __attribute__((address_space(1))) v4i_b128* gb128_t;
typedef __attribute__((address_space(3))) v4i_b128* lb128_t;
__device__ __forceinline__ void async_copy_b128(const void* g, void* l) {
  __builtin_amdgcn_global_load_async_to_lds_b128((gb128_t)(void*)g, (lb128_t)l, 0, 0);
}
#endif

// Inline erf (Abramowitz-Stegun 7.1.26, |err| < 1.5e-7): pure fma + v_exp_f32,
// avoids the __ocml_erf_f32 call that forced accumulator spills to scratch.
__device__ __forceinline__ float erf_inline(float x) {
  const float ax = fabsf(x);
  const float t  = 1.0f / fmaf(0.3275911f, ax, 1.0f);
  float p = fmaf(1.061405429f, t, -1.453152027f);
  p = fmaf(p, t, 1.421413741f);
  p = fmaf(p, t, -0.284496736f);
  p = fmaf(p, t, 0.254829592f);
  const float y = 1.0f - p * t * __expf(-ax * ax);
  return copysignf(y, x);
}
__device__ __forceinline__ float gelu_f(float x) {
  return 0.5f * x * (1.0f + erf_inline(x * 0.70710678118654752440f));
}

// ---------------------------------------------------------------------------
// LayerNorm + cast-to-bf16 (one 256-thread block per row, D multiple of 256)
// ---------------------------------------------------------------------------
__global__ __launch_bounds__(256) void ln_cast_kernel(
    const float* __restrict__ x, const float* __restrict__ gamma,
    const float* __restrict__ beta, __bf16* __restrict__ xbf,
    __bf16* __restrict__ xnbf, int D) {
  const int row = blockIdx.x;
  const int tid = threadIdx.x;
  const int per = D >> 8;  // elements per thread (4 for D=1024)
  const float* xr = x + (size_t)row * D;
  __shared__ float red[256];

  float vals[8];
  float s = 0.f;
  for (int u = 0; u < per; ++u) { vals[u] = xr[tid + (u << 8)]; s += vals[u]; }
  red[tid] = s; __syncthreads();
  for (int k = 128; k > 0; k >>= 1) { if (tid < k) red[tid] += red[tid + k]; __syncthreads(); }
  const float mu = red[0] / (float)D;
  __syncthreads();

  float v = 0.f;
  for (int u = 0; u < per; ++u) { float d = vals[u] - mu; v += d * d; }
  red[tid] = v; __syncthreads();
  for (int k = 128; k > 0; k >>= 1) { if (tid < k) red[tid] += red[tid + k]; __syncthreads(); }
  const float rstd = rsqrtf(red[0] / (float)D + 1e-5f);
  __syncthreads();

  for (int u = 0; u < per; ++u) {
    const int j = tid + (u << 8);
    const float xn = (vals[u] - mu) * rstd * gamma[j] + beta[j];
    xnbf[(size_t)row * D + j] = (__bf16)xn;
    xbf [(size_t)row * D + j] = (__bf16)vals[u];
  }
}

// ---------------------------------------------------------------------------
// Elementwise f32 -> bf16 cast (grid-stride)
// ---------------------------------------------------------------------------
__global__ __launch_bounds__(256) void cast_bf16_kernel(
    const float* __restrict__ in, __bf16* __restrict__ out, size_t n) {
  size_t i = (size_t)blockIdx.x * 256 + threadIdx.x;
  const size_t stride = (size_t)gridDim.x * 256;
  for (; i < n; i += stride) out[i] = (__bf16)in[i];
}

// ---------------------------------------------------------------------------
// Exact per-row top-k -> 0/1 byte mask.  Radix-select on order-preserving
// uint keys; ties broken by lowest index (matches jax.lax.top_k).
// One block per row; F <= 4096, multiple of 256.
// ---------------------------------------------------------------------------
__global__ __launch_bounds__(256) void topk_mask_kernel(
    const float* __restrict__ scores, const int* __restrict__ topk_ptr,
    unsigned char* __restrict__ mask, int F) {
  const int row = blockIdx.x;
  const int tid = threadIdx.x;
  __shared__ unsigned int keys[4096];
  __shared__ int red[256];

  const float* srow = scores + (size_t)row * F;
  for (int j = tid; j < F; j += 256) {
    unsigned int b = __float_as_uint(srow[j]);
    keys[j] = (b & 0x80000000u) ? ~b : (b | 0x80000000u);  // monotone key
  }
  __syncthreads();

  const int need = topk_ptr[0];
  const int per  = F >> 8;        // contiguous chunk per thread
  const int base = tid * per;

  // thr = need-th largest key value (bitwise binary search)
  unsigned int thr = 0;
  for (int bit = 31; bit >= 0; --bit) {
    const unsigned int trial = thr | (1u << bit);
    int c = 0;
    for (int u = 0; u < per; ++u) c += (keys[base + u] >= trial) ? 1 : 0;
    red[tid] = c; __syncthreads();
    for (int s2 = 128; s2 > 0; s2 >>= 1) { if (tid < s2) red[tid] += red[tid + s2]; __syncthreads(); }
    const int total = red[0];
    __syncthreads();
    if (total >= need) thr = trial;
  }

  // counts of strictly-greater and equal
  int cgt = 0, ceq = 0;
  for (int u = 0; u < per; ++u) {
    const unsigned int k = keys[base + u];
    cgt += (k > thr) ? 1 : 0;
    ceq += (k == thr) ? 1 : 0;
  }
  red[tid] = cgt; __syncthreads();
  for (int s2 = 128; s2 > 0; s2 >>= 1) { if (tid < s2) red[tid] += red[tid + s2]; __syncthreads(); }
  const int totalGt = red[0];
  __syncthreads();
  const int rem = need - totalGt;  // #ties to accept, lowest index first

  // exclusive prefix of equal-counts across threads (index order)
  red[tid] = ceq; __syncthreads();
  for (int offd = 1; offd < 256; offd <<= 1) {
    int v = (tid >= offd) ? red[tid - offd] : 0;
    __syncthreads();
    if (tid >= offd) red[tid] += v;
    __syncthreads();
  }
  int rank = red[tid] - ceq;

  unsigned char* mrow = mask + (size_t)row * F;
  for (int u = 0; u < per; ++u) {
    const unsigned int k = keys[base + u];
    unsigned char m = 0;
    if (k > thr) m = 1;
    else if (k == thr) { m = (rank < rem) ? 1 : 0; ++rank; }
    mrow[base + u] = m;
  }
}

// ---------------------------------------------------------------------------
// bf16 WMMA GEMM:  C[M,Nc] = epilogue( A[M,K] @ B[Nc,K]^T )
// A, B row-major bf16; 128x128x32 block tile, 8 waves, wave tile 32x64.
// LDS rows: 16 words (32 bf16); 16B groups XOR-swizzled by (row>>2)&3 so
// b128 stores stay aligned and fragment reads are bank-conflict-free.
// MODE 0: C bf16 = gelu(acc)
// MODE 1: C f32  = acc
// MODE 2: C bf16 = mask ? gelu(acc) : 0
// ---------------------------------------------------------------------------
template <int MODE>
__global__ __launch_bounds__(256) void gemm_wmma_kernel(
    const __bf16* __restrict__ A, const __bf16* __restrict__ B,
    void* __restrict__ Cv, const unsigned char* __restrict__ mask,
    int M, int Nc, int K) {
  constexpr int TM = 128, TN = 128, TK = 32;
  constexpr int RW = TK / 2;  // 16 uint words per row (64 B)
  __shared__ unsigned int As[2][TM * RW];
  __shared__ unsigned int Bs[2][TN * RW];

  const int tid  = threadIdx.x;
  const int lane = tid & 31;
  const int wave = tid >> 5;
  const int wm   = wave & 3;   // wave M slot: 4 * 32 rows
  const int wn   = wave >> 2;  // wave N slot: 2 * 64 cols
  const int l15  = lane & 15;
  const int half = lane >> 4;

  const int rowBlock = blockIdx.y * TM;
  const int colBlock = blockIdx.x * TN;

  // cooperative tile loads: 2 threads per row, 2 x 16B groups each
  const int ldRow = tid >> 1;
  const int g0    = (tid & 1) * 2;          // logical groups g0, g0+1
  const int swzLd = (ldRow >> 2) & 3;
  const int p0    = ((g0)     ^ swzLd) * 4; // physical word offsets
  const int p1    = ((g0 + 1) ^ swzLd) * 4;
  const __bf16* Ag = A + (size_t)(rowBlock + ldRow) * K + g0 * 8;
  const __bf16* Bg = B + (size_t)(colBlock + ldRow) * K + g0 * 8;
  unsigned int* AsRow0 = &As[0][ldRow * RW];
  unsigned int* BsRow0 = &Bs[0][ldRow * RW];
  unsigned int* AsRow1 = &As[1][ldRow * RW];
  unsigned int* BsRow1 = &Bs[1][ldRow * RW];

  v8f acc[2][4];
  for (int i = 0; i < 2; ++i)
    for (int j = 0; j < 4; ++j)
      for (int r = 0; r < 8; ++r) acc[i][j][r] = 0.0f;

  union Frag { v16bf v; unsigned int u[8]; };
  const int kIters = K / TK;

#if USE_ASYNC_LDS
  // Issue the K-tile kt2 into LDS buffer nbuf via the async-LDS engine.
  auto issue = [&](int nbuf, int kt2) {
    const __bf16* ga = Ag + (size_t)kt2 * TK;
    const __bf16* gb = Bg + (size_t)kt2 * TK;
    unsigned int* la = nbuf ? AsRow1 : AsRow0;
    unsigned int* lb = nbuf ? BsRow1 : BsRow0;
    async_copy_b128(ga,     &la[p0]);
    async_copy_b128(ga + 8, &la[p1]);
    async_copy_b128(gb,     &lb[p0]);
    async_copy_b128(gb + 8, &lb[p1]);
  };
  issue(0, 0);
#endif

  for (int kt = 0; kt < kIters; ++kt) {
    const int buf = kt & 1;
#if USE_ASYNC_LDS
    __builtin_amdgcn_s_wait_asynccnt(0);  // my tile writes landed in LDS
    __syncthreads();                      // everyone's landed; prev reads done
    if (kt + 1 < kIters) issue(buf ^ 1, kt + 1);
#else
    const __bf16* ga = Ag + (size_t)kt * TK;
    const __bf16* gb = Bg + (size_t)kt * TK;
    if (kt + 1 < kIters) {  // global_prefetch_b8 for the next K tile
      __builtin_prefetch(ga + TK, 0, 3);
      __builtin_prefetch(gb + TK, 0, 3);
    }
    const uint4 a0 = *reinterpret_cast<const uint4*>(ga);
    const uint4 a1 = *reinterpret_cast<const uint4*>(ga + 8);
    const uint4 b0 = *reinterpret_cast<const uint4*>(gb);
    const uint4 b1 = *reinterpret_cast<const uint4*>(gb + 8);
    __syncthreads();  // previous iteration's LDS reads complete
    unsigned int* la = buf ? AsRow1 : AsRow0;
    unsigned int* lb = buf ? BsRow1 : BsRow0;
    *reinterpret_cast<uint4*>(&la[p0]) = a0;
    *reinterpret_cast<uint4*>(&la[p1]) = a1;
    *reinterpret_cast<uint4*>(&lb[p0]) = b0;
    *reinterpret_cast<uint4*>(&lb[p1]) = b1;
    __syncthreads();
#endif

    const unsigned int* asb = As[buf];
    const unsigned int* bsb = Bs[buf];

    // A fragments: 16x32 bf16 per ISA layout (lanes 0-15: K 0..7,16..23;
    // lanes 16-31: K 8..15,24..31), word v holds one K pair.
    Frag af[2];
    for (int i = 0; i < 2; ++i) {
      const int m  = wm * 32 + i * 16 + l15;
      const int bA = m * RW;
      const int sw = (m >> 2) & 3;
#pragma unroll
      for (int v = 0; v < 8; ++v) {
        const int up   = v + ((v >= 4) ? 4 : 0) + 4 * half;  // logical word
        const int phys = (((up >> 2) ^ sw) << 2) | (up & 3);
        af[i].u[v] = asb[bA + phys];
      }
    }
    // B fragments: K x 16 with lanes 0-15 K 0..7,16..23, lanes 16-31 K 8..15,24..31.
    Frag bf4[4];
    for (int j = 0; j < 4; ++j) {
      const int n  = wn * 64 + j * 16 + l15;
      const int bB = n * RW;
      const int sw = (n >> 2) & 3;
#pragma unroll
      for (int v = 0; v < 8; ++v) {
        const int wb   = v + ((v >= 4) ? 4 : 0) + 4 * half;
        const int phys = (((wb >> 2) ^ sw) << 2) | (wb & 3);
        bf4[j].u[v] = bsb[bB + phys];
      }
    }

#pragma unroll
    for (int i = 0; i < 2; ++i)
#pragma unroll
      for (int j = 0; j < 4; ++j)
        acc[i][j] = __builtin_amdgcn_wmma_f32_16x16x32_bf16(
            false, af[i].v, false, bf4[j].v, (short)0, acc[i][j], false, false);
  }

  // Epilogue: C/D layout — vgpr r, lanes 0-15: M=r, lanes 16-31: M=r+8; N=lane&15
  for (int i = 0; i < 2; ++i) {
    for (int j = 0; j < 4; ++j) {
      const int col = colBlock + wn * 64 + j * 16 + l15;
#pragma unroll
      for (int r = 0; r < 8; ++r) {
        const int row = rowBlock + wm * 32 + i * 16 + r + 8 * half;
        const float v = acc[i][j][r];
        const size_t idx = (size_t)row * Nc + col;
        if (MODE == 0) {
          ((__bf16*)Cv)[idx] = (__bf16)gelu_f(v);
        } else if (MODE == 1) {
          ((float*)Cv)[idx] = v;
        } else {
          ((__bf16*)Cv)[idx] = mask[idx] ? (__bf16)gelu_f(v) : (__bf16)0.0f;
        }
      }
    }
  }
}

// ---------------------------------------------------------------------------
extern "C" void kernel_launch(void* const* d_in, const int* in_sizes, int n_in,
                              void* d_out, int out_size, void* d_ws, size_t ws_size,
                              hipStream_t stream) {
  const float* x     = (const float*)d_in[0];
  const float* W1    = (const float*)d_in[1];
  const float* W2    = (const float*)d_in[2];
  const float* Wr1   = (const float*)d_in[3];
  // d_in[4] = W_router_2 == W1 (shared init); reuse W1's bf16 copy.
  const float* gamma = (const float*)d_in[5];
  const float* beta  = (const float*)d_in[6];
  const int*   topk  = (const int*)d_in[7];

  const int D = in_sizes[5];
  const int F = in_sizes[1] / D;
  const int N = in_sizes[0] / D;

  char* ws = (char*)d_ws;
  size_t off = 0;
  auto alloc = [&](size_t bytes) -> char* {
    char* p = ws + off;
    off += (bytes + 255) & ~(size_t)255;
    return p;
  };
  __bf16* xbf   = (__bf16*)alloc((size_t)N * D * 2);
  __bf16* xnbf  = (__bf16*)alloc((size_t)N * D * 2);
  __bf16* hbf   = (__bf16*)alloc((size_t)N * D * 2);
  __bf16* w1b   = (__bf16*)alloc((size_t)F * D * 2);
  __bf16* w2b   = (__bf16*)alloc((size_t)D * F * 2);
  __bf16* wr1b  = (__bf16*)alloc((size_t)D * D * 2);
  float*  scores = (float*)alloc((size_t)N * F * 4);
  unsigned char* mask = (unsigned char*)alloc((size_t)N * F);
  __bf16* sbuf = (__bf16*)scores;  // overlay: scores dead after top-k

  // 1) weight casts + LN/cast of activations
  cast_bf16_kernel<<<2048, 256, 0, stream>>>(W1,  w1b,  (size_t)F * D);
  cast_bf16_kernel<<<2048, 256, 0, stream>>>(W2,  w2b,  (size_t)D * F);
  cast_bf16_kernel<<<1024, 256, 0, stream>>>(Wr1, wr1b, (size_t)D * D);
  ln_cast_kernel<<<N, 256, 0, stream>>>(x, gamma, beta, xbf, xnbf, D);

  // 2) h = gelu(xn @ Wr1^T)           [N, D] bf16
  gemm_wmma_kernel<0><<<dim3(D / 128, N / 128), 256, 0, stream>>>(
      xnbf, wr1b, (void*)hbf, nullptr, N, D, D);

  // 3) scores = h @ W1^T              [N, F] f32
  gemm_wmma_kernel<1><<<dim3(F / 128, N / 128), 256, 0, stream>>>(
      hbf, w1b, (void*)scores, nullptr, N, F, D);

  // 4) per-row top-k -> mask
  topk_mask_kernel<<<N, 256, 0, stream>>>(scores, topk, mask, F);

  // 5) s = gelu(x @ W1^T) * mask      [N, F] bf16  (overwrites scores region)
  gemm_wmma_kernel<2><<<dim3(F / 128, N / 128), 256, 0, stream>>>(
      xbf, w1b, (void*)sbuf, mask, N, F, D);

  // 6) out = s @ W2^T                 [N, D] f32
  gemm_wmma_kernel<1><<<dim3(D / 128, N / 128), 256, 0, stream>>>(
      sbuf, w2b, d_out, nullptr, N, D, F);
}
